// AttentionFreeTransformer_13365938226129
// MI455X (gfx1250) — matrix operand order repaired and checked by reading, not compiled, not run on hardware
//
#include <hip/hip_runtime.h>
#include <hip/hip_bf16.h>

typedef __attribute__((ext_vector_type(16))) _Float16 v16h;
typedef __attribute__((ext_vector_type(8)))  _Float16 h8;
typedef __attribute__((ext_vector_type(8)))  float    v8f;

#define BB   32
#define NN   196
#define CC   768
#define NP   (BB * NN)        // 6272 = 392*16
#define IMG  224
#define PATCH 16
#define WIN  16
#define KC   (CC * WIN)       // 12288
#define D3   (3 * CC)         // 2304
#define EPS  1e-5f

#if __has_builtin(__builtin_amdgcn_global_load_async_to_lds_b32) && \
    __has_builtin(__builtin_amdgcn_s_wait_asynccnt)
#define HAVE_ASYNC_LDS 1
typedef __attribute__((address_space(1))) int g_int;
typedef __attribute__((address_space(3))) int l_int;
#else
#define HAVE_ASYNC_LDS 0
#endif

// ---------------------------------------------------------------------------
// WMMA fragment loaders (wave32, 16x16x32 f16)
// A (16xK tile, row-major source, ld in halves):
//   lane 0-15 : M=lane,    K = kb+0..15
//   lane 16-31: M=lane-16, K = kb+16..31
// -> two contiguous 8-half (16B) vector loads per lane.
// ---------------------------------------------------------------------------
__device__ __forceinline__ v16h load_a_rowmajor(const _Float16* __restrict__ A,
                                                int row0, int ld, int kb) {
    int lane = threadIdx.x & 31;
    int half = lane >> 4;
    int m    = lane & 15;
    const _Float16* p = A + (size_t)(row0 + m) * ld + kb + 16 * half;
    h8 lo = *(const h8*)p;
    h8 hi = *(const h8*)(p + 8);
    v16h r;
#pragma unroll
    for (int e = 0; e < 8; ++e) { r[e] = lo[e]; r[e + 8] = hi[e]; }
    return r;
}

// B (Kx16 tile) where B(k, n) = W[n][k], W row-major with leading dim ld:
//   lane 0-15 : N=lane,    K = kb+0..15
//   lane 16-31: N=lane-16, K = kb+16..31
// -> one contiguous 16-half (32B) vector load per lane.
__device__ __forceinline__ v16h load_b_weight(const _Float16* __restrict__ W,
                                              int col0, int ld, int kb) {
    int lane = threadIdx.x & 31;
    int half = lane >> 4;
    int n    = lane & 15;
    const _Float16* p = W + (size_t)(col0 + n) * ld + kb + 16 * half;
    return *(const v16h*)p;
}

// B operand for the tap-unrolled conv GEMM.
// Global K index kidx = cin*16 + t  (kb is a multiple of 32).
// For this lane:  cin = kb/16 + half,  element e has tap t=e,
// source value = S[b][cin][n + e - 7] (zero outside [0,NN)).
__device__ __forceinline__ v16h load_b_conv(const _Float16* __restrict__ S,
                                            int col0, int kb) {
    int lane = threadIdx.x & 31;
    int half = lane >> 4;
    int nc   = col0 + (lane & 15);
    int b    = nc / NN;
    int n    = nc % NN;
    int cin  = (kb >> 4) + half;
    const _Float16* p = S + ((size_t)b * CC + cin) * NN + n - 7;
    v16h r;
#pragma unroll
    for (int e = 0; e < 16; ++e) {
        int src = n + e - 7;
        r[e] = (src >= 0 && src < NN) ? p[e] : (_Float16)0.f;
    }
    return r;
}

// ---------------------------------------------------------------------------
// Elementwise / setup kernels
// ---------------------------------------------------------------------------
__global__ void init_scalars(float* S) {
    if (threadIdx.x < 2 && blockIdx.x == 0) S[threadIdx.x] = 0.f;
}

__global__ void cvt_f16(const float* __restrict__ src, _Float16* __restrict__ dst, int n) {
    int i = blockIdx.x * blockDim.x + threadIdx.x;
    if (i < n) dst[i] = (_Float16)src[i];
}

// ker = exp(conv_w) - 1, laid out [cout][cin*16 + t] (conv_w already row-major (C,C,16))
__global__ void cvt_ker(const float* __restrict__ src, _Float16* __restrict__ dst, int n) {
    int i = blockIdx.x * blockDim.x + threadIdx.x;
    if (i < n) dst[i] = (_Float16)(expf(src[i]) - 1.f);
}

// im2col for the patch embedding: A[np][k], k = ci*256 + r*16 + col
__global__ void im2col_patch(const float* __restrict__ x, _Float16* __restrict__ A) {
    int i = blockIdx.x * blockDim.x + threadIdx.x;  // over NP*CC (exact)
    int np = i / CC;
    int kk = i % CC;
    int b  = np / NN;
    int pi = np % NN;
    int ph = pi / 14, pw = pi % 14;
    int ci = kk >> 8;
    int rr = (kk >> 4) & 15;
    int cl = kk & 15;
    A[i] = (_Float16)x[(((size_t)b * 3 + ci) * IMG + (ph * PATCH + rr)) * IMG + (pw * PATCH + cl)];
}

// ---------------------------------------------------------------------------
// GEMM1: patch embed.  e[np][c] = sum_k A[np][k] * Wp[c][k] + bias[c]
// Each wave: 16 rows x 64 cols. All fragment loads issued before the WMMA
// chain so the scheduler can pipeline with partial s_wait_loadcnt.
// ---------------------------------------------------------------------------
__global__ void gemm_patch(const _Float16* __restrict__ A, const _Float16* __restrict__ W,
                           const float* __restrict__ bias,
                           float* __restrict__ e32, _Float16* __restrict__ e16) {
    int wid = blockIdx.x * 8 + (threadIdx.x >> 5);   // 392 * 12 wave-tiles
    int mt = wid / (CC / 64);
    int ng = wid % (CC / 64);
    int m0 = mt * 16, n0 = ng * 64;
    v8f acc[4] = {{}, {}, {}, {}};
    for (int kb = 0; kb < CC; kb += 32) {
        v16h a = load_a_rowmajor(A, m0, CC, kb);
        v16h b[4];
#pragma unroll
        for (int j = 0; j < 4; ++j) b[j] = load_b_weight(W, n0 + j * 16, CC, kb);
#pragma unroll
        for (int j = 0; j < 4; ++j)
            acc[j] = __builtin_amdgcn_wmma_f32_16x16x32_f16(false, a, false, b[j], (short)0, acc[j], false, false);
    }
    int lane = threadIdx.x & 31;
    int half = lane >> 4;
#pragma unroll
    for (int j = 0; j < 4; ++j) {
        int n  = n0 + j * 16 + (lane & 15);
        float bn = bias[n];
#pragma unroll
        for (int r = 0; r < 8; ++r) {
            int m = m0 + r + 8 * half;
            float v = acc[j][r] + bn;
            e32[(size_t)m * CC + n] = v;
            e16[(size_t)m * CC + n] = (_Float16)v;
        }
    }
}

// ---------------------------------------------------------------------------
// GEMM2: qkv[np][d] = sum_c e[np][c] * w_qkv[d][c]; scatter q/k/v into (B,C,N)
// Each wave: 16 rows x 64 cols.
// ---------------------------------------------------------------------------
__global__ void gemm_qkv(const _Float16* __restrict__ E, const _Float16* __restrict__ W,
                         float* __restrict__ q32, _Float16* __restrict__ k16,
                         _Float16* __restrict__ v16) {
    int wid = blockIdx.x * 8 + (threadIdx.x >> 5);   // 392 * 36 wave-tiles
    int mt = wid / (D3 / 64);
    int ng = wid % (D3 / 64);
    int m0 = mt * 16, n0 = ng * 64;
    v8f acc[4] = {{}, {}, {}, {}};
    for (int kb = 0; kb < CC; kb += 32) {
        v16h a = load_a_rowmajor(E, m0, CC, kb);
        v16h b[4];
#pragma unroll
        for (int j = 0; j < 4; ++j) b[j] = load_b_weight(W, n0 + j * 16, CC, kb);
#pragma unroll
        for (int j = 0; j < 4; ++j)
            acc[j] = __builtin_amdgcn_wmma_f32_16x16x32_f16(false, a, false, b[j], (short)0, acc[j], false, false);
    }
    int lane = threadIdx.x & 31;
    int half = lane >> 4;
#pragma unroll
    for (int j = 0; j < 4; ++j) {
        int d = n0 + j * 16 + (lane & 15);
#pragma unroll
        for (int r = 0; r < 8; ++r) {
            int np = m0 + r + 8 * half;
            int b  = np / NN;
            int n  = np % NN;
            float v = acc[j][r];
            if (d < CC)            q32[((size_t)b * CC + d)          * NN + n] = v;
            else if (d < 2 * CC)   k16[((size_t)b * CC + (d - CC))   * NN + n] = (_Float16)v;
            else                   v16[((size_t)b * CC + (d - 2*CC)) * NN + n] = (_Float16)v;
        }
    }
}

// ---------------------------------------------------------------------------
// kv = k*v (f16) and global reductions S1 = sum(exp(k)*v), S2 = sum(exp(k))
// ---------------------------------------------------------------------------
__global__ void reduce_kv(const _Float16* __restrict__ K, const _Float16* __restrict__ V,
                          _Float16* __restrict__ KV, float* __restrict__ S) {
    int i = blockIdx.x * 256 + threadIdx.x;          // NP*CC exact
    float kf = (float)K[i];
    float vf = (float)V[i];
    KV[i] = (_Float16)(kf * vf);
    float ek = expf(kf);
    __shared__ float sh0[256], sh1[256];
    sh0[threadIdx.x] = ek * vf;
    sh1[threadIdx.x] = ek;
    __syncthreads();
    for (int s = 128; s > 0; s >>= 1) {
        if (threadIdx.x < s) {
            sh0[threadIdx.x] += sh0[threadIdx.x + s];
            sh1[threadIdx.x] += sh1[threadIdx.x + s];
        }
        __syncthreads();
    }
    if (threadIdx.x == 0) {
        atomicAdd(&S[0], sh0[0]);
        atomicAdd(&S[1], sh1[0]);
    }
}

// ---------------------------------------------------------------------------
// GEMM3: dual conv GEMM over K' = cin*16+t (12288).
// Each wave: 64 couts x 16 cols; expensive B-frags (kv, k) reused across 4
// m-tiles. All loads of a K-step issued before the 8-WMMA chain.
// Epilogue: a_pre[np][cout] = sigmoid(q) * (num+S1) / (den+S2)
// ---------------------------------------------------------------------------
__global__ void gemm_conv(const _Float16* __restrict__ KER, const _Float16* __restrict__ KV,
                          const _Float16* __restrict__ Kf, const float* __restrict__ q32,
                          const float* __restrict__ S, float* __restrict__ apre) {
    int wid = blockIdx.x * 8 + (threadIdx.x >> 5);   // 12 * 392 wave-tiles
    int mg = wid / (NP / 16);
    int nt = wid % (NP / 16);
    int m0 = mg * 64, n0 = nt * 16;
    v8f accn[4] = {{}, {}, {}, {}};
    v8f accd[4] = {{}, {}, {}, {}};
    for (int kb = 0; kb < KC; kb += 32) {
        v16h bkv = load_b_conv(KV, n0, kb);
        v16h bk  = load_b_conv(Kf, n0, kb);
        v16h a[4];
#pragma unroll
        for (int j = 0; j < 4; ++j) {
            __builtin_prefetch(KER + (size_t)(m0 + j * 16) * KC + kb + 256, 0, 1);
            a[j] = load_a_rowmajor(KER, m0 + j * 16, KC, kb);
        }
#pragma unroll
        for (int j = 0; j < 4; ++j) {
            accn[j] = __builtin_amdgcn_wmma_f32_16x16x32_f16(false, a[j], false, bkv, (short)0, accn[j], false, false);
            accd[j] = __builtin_amdgcn_wmma_f32_16x16x32_f16(false, a[j], false, bk,  (short)0, accd[j], false, false);
        }
    }
    float s1 = S[0], s2 = S[1];
    int lane = threadIdx.x & 31;
    int half = lane >> 4;
    int nc   = n0 + (lane & 15);
    int b    = nc / NN;
    int n    = nc % NN;
#pragma unroll
    for (int j = 0; j < 4; ++j) {
#pragma unroll
        for (int r = 0; r < 8; ++r) {
            int cout = m0 + j * 16 + r + 8 * half;
            float qv  = q32[((size_t)b * CC + cout) * NN + n];
            float sig = 1.f / (1.f + expf(-qv));
            float av  = sig * (accn[j][r] + s1) / (accd[j][r] + s2);
            apre[((size_t)b * NN + n) * CC + cout] = av;
        }
    }
}

// ---------------------------------------------------------------------------
// LayerNorm over C + residual:  out[np][c] = e + (a-mu)*rsqrt(var+eps)*gamma + beta
// Stages the a_pre row into LDS with gfx1250 async-to-LDS loads when available.
// ---------------------------------------------------------------------------
__global__ void ln_residual(const float* __restrict__ apre, const float* __restrict__ e32,
                            const float* __restrict__ gamma, const float* __restrict__ beta,
                            float* __restrict__ out) {
    int np = blockIdx.x;
    const float* row = apre + (size_t)np * CC;
    __shared__ float arow[CC];

#if HAVE_ASYNC_LDS
    {
#pragma unroll
        for (int j = 0; j < 3; ++j) {
            int c = threadIdx.x + j * 256;
            g_int* gsrc = (g_int*)(unsigned long long)(row + c);
            l_int* ldst = (l_int*)(unsigned)(unsigned long long)(arow + c);
            __builtin_amdgcn_global_load_async_to_lds_b32(gsrc, ldst, 0, 0);
        }
        __builtin_amdgcn_s_wait_asynccnt(0);
    }
    __syncthreads();
#else
#pragma unroll
    for (int j = 0; j < 3; ++j) {
        int c = threadIdx.x + j * 256;
        arow[c] = row[c];
    }
    __syncthreads();
#endif

    float vals[3], s = 0.f, s2 = 0.f;
#pragma unroll
    for (int j = 0; j < 3; ++j) {
        float v = arow[threadIdx.x + j * 256];
        vals[j] = v;
        s  += v;
        s2 += v * v;
    }
    __shared__ float sh0[256], sh1[256];
    sh0[threadIdx.x] = s;
    sh1[threadIdx.x] = s2;
    __syncthreads();
    for (int t = 128; t > 0; t >>= 1) {
        if (threadIdx.x < t) {
            sh0[threadIdx.x] += sh0[threadIdx.x + t];
            sh1[threadIdx.x] += sh1[threadIdx.x + t];
        }
        __syncthreads();
    }
    __shared__ float mu_s, ri_s;
    if (threadIdx.x == 0) {
        float mu  = sh0[0] / (float)CC;
        float var = sh1[0] / (float)CC - mu * mu;
        mu_s = mu;
        ri_s = rsqrtf(var + EPS);
    }
    __syncthreads();
    float mu = mu_s, ri = ri_s;
#pragma unroll
    for (int j = 0; j < 3; ++j) {
        int c = threadIdx.x + j * 256;
        out[(size_t)np * CC + c] = e32[(size_t)np * CC + c] +
                                   (vals[j] - mu) * ri * gamma[c] + beta[c];
    }
}

// ---------------------------------------------------------------------------
extern "C" void kernel_launch(void* const* d_in, const int* in_sizes, int n_in,
                              void* d_out, int out_size, void* d_ws, size_t ws_size,
                              hipStream_t stream) {
    const float* x       = (const float*)d_in[0];
    const float* patch_w = (const float*)d_in[1];
    const float* patch_b = (const float*)d_in[2];
    const float* w_qkv   = (const float*)d_in[3];
    const float* conv_w  = (const float*)d_in[4];
    const float* gamma   = (const float*)d_in[5];
    const float* beta    = (const float*)d_in[6];
    float* out = (float*)d_out;

    // workspace layout (all section sizes are multiples of 256B)
    char* ws = (char*)d_ws;
    size_t off = 0;
    _Float16* A16   = (_Float16*)(ws + off); off += (size_t)NP * CC * 2;  // im2col
    _Float16* WP16  = (_Float16*)(ws + off); off += (size_t)CC * CC * 2;
    _Float16* WQ16  = (_Float16*)(ws + off); off += (size_t)D3 * CC * 2;
    float*    E32   = (float*)   (ws + off); off += (size_t)NP * CC * 4;
    _Float16* E16   = (_Float16*)(ws + off); off += (size_t)NP * CC * 2;
    float*    Q32   = (float*)   (ws + off); off += (size_t)NP * CC * 4;  // (B,C,N)
    _Float16* K16   = (_Float16*)(ws + off); off += (size_t)NP * CC * 2;  // (B,C,N)
    _Float16* V16   = (_Float16*)(ws + off); off += (size_t)NP * CC * 2;  // (B,C,N)
    _Float16* KV16  = (_Float16*)(ws + off); off += (size_t)NP * CC * 2;  // (B,C,N)
    _Float16* KER16 = (_Float16*)(ws + off); off += (size_t)CC * KC * 2;  // [cout][cin*16+t]
    float*    APRE  = (float*)   (ws + off); off += (size_t)NP * CC * 4;  // (B,N,C)
    float*    SCAL  = (float*)   (ws + off); off += 256;                  // S1, S2

    init_scalars<<<1, 64, 0, stream>>>(SCAL);
    im2col_patch<<<(NP * CC) / 256, 256, 0, stream>>>(x, A16);
    cvt_f16<<<(CC * CC) / 256, 256, 0, stream>>>(patch_w, WP16, CC * CC);
    cvt_f16<<<(D3 * CC) / 256, 256, 0, stream>>>(w_qkv, WQ16, D3 * CC);
    cvt_ker<<<(CC * KC) / 256, 256, 0, stream>>>(conv_w, KER16, CC * KC);

    gemm_patch<<<(392 * 12) / 8, 256, 0, stream>>>(A16, WP16, patch_b, E32, E16);
    gemm_qkv<<<(392 * 36) / 8, 256, 0, stream>>>(E16, WQ16, Q32, K16, V16);
    reduce_kv<<<(NP * CC) / 256, 256, 0, stream>>>(K16, V16, KV16, SCAL);
    gemm_conv<<<(12 * 392) / 8, 256, 0, stream>>>(KER16, KV16, K16, Q32, SCAL, APRE);
    ln_residual<<<NP, 256, 0, stream>>>(APRE, E32, gamma, beta, out);
}
